// UNet20_69011534512135
// MI455X (gfx1250) — matrix operand order repaired and checked by reading, not compiled, not run on hardware
//
#include <hip/hip_runtime.h>

typedef __attribute__((ext_vector_type(16))) _Float16 v16h;
typedef __attribute__((ext_vector_type(8)))  _Float16 v8h;
typedef __attribute__((ext_vector_type(8)))  float    v8f;

#define TPB 256

// ---------------------------------------------------------------------------
// Implicit-GEMM conv via WMMA f16 (D = sign(W) [Cout x K] * im2col(act) [K x Npix])
// Block: 256 threads = 8 wave32 waves. Block tile: 64 Cout x 128 pixels,
// K step 32. Each wave: 16 pixels, 4 M-subtiles -> 4 WMMAs per K-step with a
// single shared B fragment (register blocking on M).
// Supports ks=3 (3x3 VALID conv) and ks=1 (1x1 conv / transposed-conv subpixel
// with strided output: out(y,x) -> (oy0 + y*ostep, ox0 + x*ostep), channel
// offset oc0 inside a larger outC-channel tensor (for concat writes).
// ---------------------------------------------------------------------------
__global__ __launch_bounds__(256) void k_conv_wmma(
    const _Float16* __restrict__ act, const _Float16* __restrict__ wgt,
    float* __restrict__ out,
    int NB, int Cin, int Hin, int Win, int Cout, int ks,
    int outC, int outH, int outW, int oc0, int oy0, int ox0, int ostep)
{
  const int Hc   = Hin - (ks - 1);
  const int Wc   = Win - (ks - 1);
  const int K    = Cin * ks * ks;
  const int Kpad = (K + 31) & ~31;
  const int Npix = NB * Hc * Wc;
  const int tid  = threadIdx.x;
  const int lane = tid & 31;
  const int wave = tid >> 5;
  const int pix0 = blockIdx.x * 128;
  const int m0   = blockIdx.y * 64;
  const int HWin = Hin * Win;
  const int hw   = Hc * Wc;

  // Row stride 40 halves = 80 B (multiple of 16 B -> aligned b128 reads,
  // not a multiple of 256 B -> banks spread).
  __shared__ __align__(16) _Float16 As[64][40];   // weights:     [m][k]
  __shared__ __align__(16) _Float16 Bs[128][40];  // activations: [pixel][k]

  // Per-thread base addresses of the 16 B-tile pixels this thread fills.
  // i = tid + 256*j  =>  k_local = lane (fixed),  p_local = wave + 8*j.
  int bad[16];
#pragma unroll
  for (int j = 0; j < 16; ++j) {
    int p = pix0 + wave + 8 * j;
    int b = -1;
    if (p < Npix) {
      int n = p / hw, rem = p - n * hw;
      int yy = rem / Wc, xx = rem - yy * Wc;
      b = (n * Cin * Hin + yy) * Win + xx;
    }
    bad[j] = b;
  }

  v8f acc[4] = {{}, {}, {}, {}};
  const int kbA  = (lane >= 16) ? 8 : 0;    // A frag: K 0-7/16-23 vs 8-15/24-31
  const int kbB  = (lane >= 16) ? 8 : 0;    // B frag: K 0-7/16-23 vs 8-15/24-31
  const int mrow = lane & 15;
  const int ncol = wave * 16 + (lane & 15);
  const int kss  = ks * ks;

  for (int kk = 0; kk < Kpad; kk += 32) {
    // --- stage A tile (64 x 32 weights), 8 elems/thread ---
#pragma unroll
    for (int i = tid; i < 2048; i += 256) {
      int ml = i >> 5, kl = i & 31;
      int m  = m0 + ml;
      _Float16 v = (_Float16)0.f;
      if (m < Cout) v = wgt[(size_t)m * Kpad + kk + kl];
      As[ml][kl] = v;
    }
    if (kk + 32 < Kpad)  // gfx1250 global_prefetch_b8 for next weight tile
      __builtin_prefetch(wgt + (size_t)(m0 + mrow) * Kpad + kk + 32, 0, 0);

    // --- stage B tile (128 pixels x 32 K), implicit im2col gather ---
    int  k  = kk + lane;
    bool kv = (k < K);
    int  koff = 0;
    if (kv) {
      int c = k / kss, r = k - c * kss;
      int dy = r / ks, dx = r - dy * ks;
      koff = c * HWin + dy * Win + dx;
    }
#pragma unroll
    for (int j = 0; j < 16; ++j) {
      _Float16 v = (_Float16)0.f;
      if (kv && bad[j] >= 0) v = act[(size_t)(bad[j] + koff)];
      Bs[wave + 8 * j][lane] = v;   // row = pixel, col = k (lane): coalesced
    }
    __syncthreads();

    // --- B fragment: two contiguous b128 LDS reads ---
    v8h b0 = *(const v8h*)&Bs[ncol][kbB];
    v8h b1 = *(const v8h*)&Bs[ncol][kbB + 16];
    v16h b;
#pragma unroll
    for (int j = 0; j < 8; ++j) { b[j] = b0[j]; b[j + 8] = b1[j]; }

    // --- 4 M-subtiles reuse the B fragment: 4 WMMAs per K-step ---
#pragma unroll
    for (int mt = 0; mt < 4; ++mt) {
      v8h a0 = *(const v8h*)&As[mt * 16 + mrow][kbA];
      v8h a1 = *(const v8h*)&As[mt * 16 + mrow][kbA + 16];
      v16h a;
#pragma unroll
      for (int j = 0; j < 8; ++j) { a[j] = a0[j]; a[j + 8] = a1[j]; }
      acc[mt] = __builtin_amdgcn_wmma_f32_16x16x32_f16(
          false, a, false, b, (short)0, acc[mt], false, false);
    }
    __syncthreads();
  }

  // --- epilogue: C layout = VGPR r -> M=r (lanes 0-15) / M=r+8 (lanes 16-31)
  int p = pix0 + wave * 16 + (lane & 15);
  if (p < Npix) {
    int n = p / hw, rem = p - n * hw;
    int yy = rem / Wc, xx = rem - yy * Wc;
    int oy = oy0 + yy * ostep, ox = ox0 + xx * ostep;
#pragma unroll
    for (int mt = 0; mt < 4; ++mt) {
      int mb = m0 + mt * 16 + ((lane >= 16) ? 8 : 0);
#pragma unroll
      for (int r = 0; r < 8; ++r) {
        int m = mb + r;
        if (m < Cout)
          out[((size_t)(n * outC + oc0 + m) * outH + oy) * outW + ox] = acc[mt][r];
      }
    }
  }
}

// --- BN statistics: one block per channel, mean + rsqrt(var+eps) -------------
__global__ __launch_bounds__(256) void k_bn_stats(
    const float* __restrict__ x, float* __restrict__ mean, float* __restrict__ rstd,
    int NB, int C, int HW)
{
  int c = blockIdx.x;
  int cnt = NB * HW;
  float s = 0.f, s2 = 0.f;
  for (int i = threadIdx.x; i < cnt; i += 256) {
    int n = i / HW, r = i - n * HW;
    float v = x[((size_t)n * C + c) * HW + r];
    s += v; s2 += v * v;
  }
  __shared__ float sb[256], qb[256];
  sb[threadIdx.x] = s; qb[threadIdx.x] = s2;
  __syncthreads();
  for (int o = 128; o; o >>= 1) {
    if (threadIdx.x < o) {
      sb[threadIdx.x] += sb[threadIdx.x + o];
      qb[threadIdx.x] += qb[threadIdx.x + o];
    }
    __syncthreads();
  }
  if (threadIdx.x == 0) {
    float m = sb[0] / (float)cnt;
    float v = qb[0] / (float)cnt - m * m;
    mean[c] = m;
    rstd[c] = rsqrtf(v + 1e-5f);
  }
}

// --- BN apply (+optional ReLU), output f16 activation or f32 ----------------
__global__ __launch_bounds__(256) void k_bn_apply(
    const float* __restrict__ x, const float* __restrict__ mean,
    const float* __restrict__ rstd, _Float16* __restrict__ oh,
    float* __restrict__ of, int C, int HW, int total, int relu)
{
  int i = blockIdx.x * 256 + threadIdx.x;
  if (i >= total) return;
  int c = (i / HW) % C;
  float v = (x[i] - mean[c]) * rstd[c];
  if (relu) v = fmaxf(v, 0.f);
  if (of) of[i] = v;
  else    oh[i] = (_Float16)v;
}

// --- 2x2 maxpool stride 2 ---------------------------------------------------
__global__ __launch_bounds__(256) void k_pool(
    const float* __restrict__ x, float* __restrict__ y, int NC, int H, int W)
{
  int Ho = H >> 1, Wo = W >> 1;
  int total = NC * Ho * Wo;
  int i = blockIdx.x * 256 + threadIdx.x;
  if (i >= total) return;
  int nc = i / (Ho * Wo), r = i - nc * (Ho * Wo);
  int yo = r / Wo, xo = r - yo * Wo;
  const float* p = x + ((size_t)nc * H + 2 * yo) * W + 2 * xo;
  y[i] = fmaxf(fmaxf(p[0], p[1]), fmaxf(p[W], p[W + 1]));
}

// --- crop-copy / channel-offset copy (crops and concat halves) --------------
__global__ __launch_bounds__(256) void k_copy_crop(
    const float* __restrict__ x, float* __restrict__ dst,
    int NB, int C, int H, int W, int crop, int dstC, int dstc0)
{
  int Ho = H - 2 * crop, Wo = W - 2 * crop;
  int total = NB * C * Ho * Wo;
  int i = blockIdx.x * 256 + threadIdx.x;
  if (i >= total) return;
  int n = i / (C * Ho * Wo), r = i - n * (C * Ho * Wo);
  int c = r / (Ho * Wo); r -= c * (Ho * Wo);
  int y = r / Wo, xw = r - y * Wo;
  dst[((size_t)(n * dstC + dstc0 + c) * Ho + y) * Wo + xw] =
      x[((size_t)(n * C + c) * H + (y + crop)) * W + (xw + crop)];
}

// --- f32 -> f16 -------------------------------------------------------------
__global__ __launch_bounds__(256) void k_cvt(
    const float* __restrict__ x, _Float16* __restrict__ y, int n)
{
  int i = blockIdx.x * 256 + threadIdx.x;
  if (i < n) y[i] = (_Float16)x[i];
}

// --- sign(w) -> f16, rows padded to Kpad ------------------------------------
__global__ __launch_bounds__(256) void k_signw(
    const float* __restrict__ w, _Float16* __restrict__ o,
    int Cout, int K, int Kpad)
{
  int total = Cout * Kpad;
  int i = blockIdx.x * 256 + threadIdx.x;
  if (i >= total) return;
  int m = i / Kpad, k = i - m * Kpad;
  float v = 0.f;
  if (k < K) {
    float t = w[(size_t)m * K + k];
    v = (t > 0.f) ? 1.f : ((t < 0.f) ? -1.f : 0.f);
  }
  o[i] = (_Float16)v;
}

// --- sign(u[c][o][p][q]) -> 4 matrices [pq][O][Cpad] -------------------------
__global__ __launch_bounds__(256) void k_signu(
    const float* __restrict__ u, _Float16* __restrict__ o,
    int C, int O, int Cpad)
{
  int total = 4 * O * Cpad;
  int i = blockIdx.x * 256 + threadIdx.x;
  if (i >= total) return;
  int pq = i / (O * Cpad), r = i - pq * (O * Cpad);
  int oc = r / Cpad, c = r - oc * Cpad;
  float v = 0.f;
  if (c < C) {
    float t = u[(((size_t)c * O + oc) << 2) + pq];
    v = (t > 0.f) ? 1.f : ((t < 0.f) ? -1.f : 0.f);
  }
  o[i] = (_Float16)v;
}

static inline int cdiv(int a, int b) { return (a + b - 1) / b; }

extern "C" void kernel_launch(void* const* d_in, const int* in_sizes, int n_in,
                              void* d_out, int out_size, void* d_ws, size_t ws_size,
                              hipStream_t stream)
{
  (void)in_sizes; (void)n_in; (void)out_size; (void)ws_size;
  const float* X = (const float*)d_in[0];
  const float* Wf[20];
  for (int i = 1; i <= 19; ++i) Wf[i] = (const float*)d_in[i];
  const float* Uf[4];
  for (int i = 0; i < 4; ++i) Uf[i] = (const float*)d_in[20 + i];

  static const int CO[20]  = {0,64,64,128,128,256,256,512,512,1024,1024,512,512,256,256,128,128,64,64,2};
  static const int CI[20]  = {0,1,64,64,128,128,256,256,512,512,1024,1024,512,512,256,256,128,128,64,64};
  static const int KSZ[20] = {0,3,3,3,3,3,3,3,3,3,3,3,3,3,3,3,3,3,3,1};
  static const int UC[4] = {1024,512,256,128};
  static const int UO[4] = {512,256,128,64};

  char* ws = (char*)d_ws;
  size_t off = 0;
  auto alloc = [&](size_t bytes) -> void* {
    void* p = ws + off;
    off = (off + bytes + 255) & ~(size_t)255;
    return p;
  };

  const size_t MAXE = (size_t)2 * 64 * 572 * 572;  // largest f32 tensor
  float*    F1  = (float*)alloc(MAXE * sizeof(float));
  float*    F2  = (float*)alloc(MAXE * sizeof(float));
  _Float16* ACT = (_Float16*)alloc(MAXE * sizeof(_Float16));
  float* Y1 = (float*)alloc((size_t)2 * 64  * 392 * 392 * sizeof(float));
  float* Y2 = (float*)alloc((size_t)2 * 128 * 200 * 200 * sizeof(float));
  float* Y3 = (float*)alloc((size_t)2 * 256 * 104 * 104 * sizeof(float));
  float* Y4 = (float*)alloc((size_t)2 * 512 * 56  * 56  * sizeof(float));
  float* MEAN = (float*)alloc(1024 * sizeof(float));
  float* RSTD = (float*)alloc(1024 * sizeof(float));

  // Pack sign(weights) as f16 GEMM matrices (deterministic every call).
  _Float16* wb[20];
  for (int l = 1; l <= 19; ++l) {
    int K = CI[l] * KSZ[l] * KSZ[l];
    int Kpad = (K + 31) & ~31;
    wb[l] = (_Float16*)alloc((size_t)CO[l] * Kpad * sizeof(_Float16));
    int tot = CO[l] * Kpad;
    k_signw<<<cdiv(tot, TPB), TPB, 0, stream>>>(Wf[l], wb[l], CO[l], K, Kpad);
  }
  _Float16* ub[4]; int UCP[4];
  for (int l = 0; l < 4; ++l) {
    UCP[l] = (UC[l] + 31) & ~31;
    ub[l] = (_Float16*)alloc((size_t)4 * UO[l] * UCP[l] * sizeof(_Float16));
    int tot = 4 * UO[l] * UCP[l];
    k_signu<<<cdiv(tot, TPB), TPB, 0, stream>>>(Uf[l], ub[l], UC[l], UO[l], UCP[l]);
  }

  auto conv = [&](const _Float16* a, const _Float16* w, float* o,
                  int Cin, int Hin, int Cout, int ks,
                  int outC, int outH, int oc0, int oy0, int ox0, int ostep) {
    int Hc = Hin - (ks - 1);
    int Npix = 2 * Hc * Hc;
    dim3 g(cdiv(Npix, 128), cdiv(Cout, 64));
    k_conv_wmma<<<g, TPB, 0, stream>>>(a, w, o, 2, Cin, Hin, Hin, Cout, ks,
                                       outC, outH, outH, oc0, oy0, ox0, ostep);
  };
  auto bn = [&](const float* xx, int C, int H, int relu, _Float16* oh, float* of) {
    k_bn_stats<<<C, TPB, 0, stream>>>(xx, MEAN, RSTD, 2, C, H * H);
    int tot = 2 * C * H * H;
    k_bn_apply<<<cdiv(tot, TPB), TPB, 0, stream>>>(xx, MEAN, RSTD, oh, of, C, H * H, tot, relu);
  };
  auto pool = [&](const float* xx, float* yy, int C, int H) {
    int tot = 2 * C * (H / 2) * (H / 2);
    k_pool<<<cdiv(tot, TPB), TPB, 0, stream>>>(xx, yy, 2 * C, H, H);
  };
  auto copycrop = [&](const float* xx, float* dst, int C, int H, int crop, int dstC, int dstc0) {
    int Ho = H - 2 * crop;
    int tot = 2 * C * Ho * Ho;
    k_copy_crop<<<cdiv(tot, TPB), TPB, 0, stream>>>(xx, dst, 2, C, H, H, crop, dstC, dstc0);
  };
  auto up = [&](const _Float16* a, int l, int Hin, float* dst, int dstC, int dstc0) {
    for (int pq = 0; pq < 4; ++pq)  // 4 subpixel 1x1 WMMA GEMMs, strided writes
      conv(a, ub[l] + (size_t)pq * UO[l] * UCP[l], dst,
           UC[l], Hin, UO[l], 1, dstC, 2 * Hin, dstc0, pq >> 1, pq & 1, 2);
  };

  // ---------------- encoder ----------------
  { int tot = 2 * 1 * 572 * 572; k_cvt<<<cdiv(tot, TPB), TPB, 0, stream>>>(X, ACT, tot); }
  conv(ACT, wb[1], F1, 1, 572, 64, 3, 64, 570, 0, 0, 0, 1);
  bn(F1, 64, 570, 1, ACT, nullptr);
  conv(ACT, wb[2], F1, 64, 570, 64, 3, 64, 568, 0, 0, 0, 1);
  copycrop(F1, Y1, 64, 568, 88, 64, 0);
  pool(F1, F2, 64, 568);
  bn(F2, 64, 284, 1, ACT, nullptr);
  conv(ACT, wb[3], F1, 64, 284, 128, 3, 128, 282, 0, 0, 0, 1);
  bn(F1, 128, 282, 1, ACT, nullptr);
  conv(ACT, wb[4], F1, 128, 282, 128, 3, 128, 280, 0, 0, 0, 1);
  copycrop(F1, Y2, 128, 280, 40, 128, 0);
  pool(F1, F2, 128, 280);
  bn(F2, 128, 140, 1, ACT, nullptr);
  conv(ACT, wb[5], F1, 128, 140, 256, 3, 256, 138, 0, 0, 0, 1);
  bn(F1, 256, 138, 1, ACT, nullptr);
  conv(ACT, wb[6], F1, 256, 138, 256, 3, 256, 136, 0, 0, 0, 1);
  copycrop(F1, Y3, 256, 136, 16, 256, 0);
  pool(F1, F2, 256, 136);
  bn(F2, 256, 68, 1, ACT, nullptr);
  conv(ACT, wb[7], F1, 256, 68, 512, 3, 512, 66, 0, 0, 0, 1);
  bn(F1, 512, 66, 1, ACT, nullptr);
  conv(ACT, wb[8], F1, 512, 66, 512, 3, 512, 64, 0, 0, 0, 1);
  copycrop(F1, Y4, 512, 64, 4, 512, 0);
  pool(F1, F2, 512, 64);
  bn(F2, 512, 32, 1, ACT, nullptr);
  conv(ACT, wb[9], F1, 512, 32, 1024, 3, 1024, 30, 0, 0, 0, 1);
  bn(F1, 1024, 30, 1, ACT, nullptr);
  conv(ACT, wb[10], F1, 1024, 30, 1024, 3, 1024, 28, 0, 0, 0, 1);
  bn(F1, 1024, 28, 1, ACT, nullptr);

  // ---------------- decoder ----------------
  copycrop(Y4, F1, 512, 56, 0, 1024, 0);
  up(ACT, 0, 28, F1, 1024, 512);
  bn(F1, 1024, 56, 0, ACT, nullptr);
  conv(ACT, wb[11], F1, 1024, 56, 512, 3, 512, 54, 0, 0, 0, 1);
  bn(F1, 512, 54, 1, ACT, nullptr);
  conv(ACT, wb[12], F1, 512, 54, 512, 3, 512, 52, 0, 0, 0, 1);
  bn(F1, 512, 52, 1, ACT, nullptr);

  copycrop(Y3, F1, 256, 104, 0, 512, 0);
  up(ACT, 1, 52, F1, 512, 256);
  bn(F1, 512, 104, 0, ACT, nullptr);
  conv(ACT, wb[13], F1, 512, 104, 256, 3, 256, 102, 0, 0, 0, 1);
  bn(F1, 256, 102, 1, ACT, nullptr);
  conv(ACT, wb[14], F1, 256, 102, 256, 3, 256, 100, 0, 0, 0, 1);
  bn(F1, 256, 100, 1, ACT, nullptr);

  copycrop(Y2, F1, 128, 200, 0, 256, 0);
  up(ACT, 2, 100, F1, 256, 128);
  bn(F1, 256, 200, 0, ACT, nullptr);
  conv(ACT, wb[15], F1, 256, 200, 128, 3, 128, 198, 0, 0, 0, 1);
  bn(F1, 128, 198, 1, ACT, nullptr);
  conv(ACT, wb[16], F1, 128, 198, 128, 3, 128, 196, 0, 0, 0, 1);
  bn(F1, 128, 196, 1, ACT, nullptr);

  copycrop(Y1, F1, 64, 392, 0, 128, 0);
  up(ACT, 3, 196, F1, 128, 64);
  bn(F1, 128, 392, 0, ACT, nullptr);
  conv(ACT, wb[17], F1, 128, 392, 64, 3, 64, 390, 0, 0, 0, 1);
  bn(F1, 64, 390, 1, ACT, nullptr);
  conv(ACT, wb[18], F1, 64, 390, 64, 3, 64, 388, 0, 0, 0, 1);
  bn(F1, 64, 388, 1, ACT, nullptr);
  conv(ACT, wb[19], F1, 64, 388, 2, 1, 2, 388, 0, 0, 0, 1);
  bn(F1, 2, 388, 0, nullptr, (float*)d_out);
}
